// MultiHeadSelfAttention_32435593020174
// MI455X (gfx1250) — matrix operand order repaired and checked by reading, not compiled, not run on hardware
//
#include <hip/hip_runtime.h>
#include <hip/hip_bf16.h>

typedef __attribute__((ext_vector_type(8)))  _Float16 v8h;
typedef __attribute__((ext_vector_type(16))) _Float16 v16h;
typedef __attribute__((ext_vector_type(8)))  float    v8f;

// ---------------------------------------------------------------------------
// Fragment loaders for CDNA5 wave32 WMMA f16 16x16x32
// A (16xK=32, row-major, ld in f16 elems): lane holds row (lane&15),
//   K runs [base, base+8) and [base+16, base+24), base = (lane>>4)*8.
// B (K=32 x 16, column-major i.e. K contiguous per column, ldc in elems):
//   lane holds column (lane&15), K runs [base, base+8) and [base+16, base+24), base = (lane>>4)*8.
// ---------------------------------------------------------------------------
__device__ __forceinline__ v16h load_afrag(const _Float16* p, int ld, int lane) {
    int row  = lane & 15;
    int base = (lane >> 4) * 8;
    const _Float16* r0 = p + row * ld + base;
    v8h lo = *(const v8h*)(r0);
    v8h hi = *(const v8h*)(r0 + 16);
    return __builtin_shufflevector(lo, hi, 0,1,2,3,4,5,6,7,8,9,10,11,12,13,14,15);
}

__device__ __forceinline__ v16h load_bfrag(const _Float16* p, int ldc, int lane) {
    int col = lane & 15;
    int kb  = (lane >> 4) * 8;
    const _Float16* r0 = p + col * ldc + kb;
    v8h lo = *(const v8h*)(r0);
    v8h hi = *(const v8h*)(r0 + 16);
    return __builtin_shufflevector(lo, hi, 0,1,2,3,4,5,6,7,8,9,10,11,12,13,14,15);
}

__device__ __forceinline__ v8f wmma16(v16h a, v16h b, v8f c) {
    // D = A x B + C ; f32 accumulate
    return __builtin_amdgcn_wmma_f32_16x16x32_f16(false, a, false, b, (short)0, c, false, false);
}

// ---------------------------------------------------------------------------
// Problem constants
// ---------------------------------------------------------------------------
#define BATCH 4
#define CCH   512
#define LEN   2048
#define HEADS 8
#define DH    64

// workspace layout (bytes)
#define OFF_XT  ((size_t)0)                                   // [B][L][C] f16  : 8 MB
#define OFF_WH  (OFF_XT + (size_t)BATCH*LEN*CCH*2)            // [3][C][C] f16  : 1.5 MB
#define OFF_QF  (OFF_WH + (size_t)3*CCH*CCH*2)                // [B][H][L][D] f16
#define OFF_KF  (OFF_QF + (size_t)BATCH*CCH*LEN*2)            // [B][H][L][D] f16
#define OFF_VH  (OFF_KF + (size_t)BATCH*CCH*LEN*2)            // [B][C][L]    f16

// ---------------------------------------------------------------------------
// Kernel 0a: x [B,C,L] f32  ->  xT [B,L,C] f16   (transpose for B-fragments)
// ---------------------------------------------------------------------------
__global__ __launch_bounds__(256) void cvt_x_kernel(const float* __restrict__ x,
                                                    _Float16* __restrict__ xT) {
    int i = blockIdx.x * 256 + threadIdx.x;              // over B*C*L = 2^22
    int l = i & (LEN - 1);
    int c = (i >> 11) & (CCH - 1);
    int b = i >> 20;
    xT[(((size_t)b * LEN + l) << 9) + c] = (_Float16)x[i];
}

// Kernel 0b: pack Wq|Wk|Wv -> Wh f16 [3][512][512]
__global__ __launch_bounds__(256) void cvt_w_kernel(const float* __restrict__ Wq,
                                                    const float* __restrict__ Wk,
                                                    const float* __restrict__ Wv,
                                                    _Float16* __restrict__ Wh) {
    int i = blockIdx.x * 256 + threadIdx.x;              // over 3*512*512
    int w = i >> 18;
    int j = i & (CCH * CCH - 1);
    const float* src = (w == 0) ? Wq : (w == 1) ? Wk : Wv;
    Wh[i] = (_Float16)src[j];
}

// ---------------------------------------------------------------------------
// Kernel 1: QKV projection. Per wave: 16(M=out-ch) x 64(N=seq) tile, K=512.
//   Q,K stored [B,H,L,D] (d contiguous); V stored [B,C,L] (l contiguous).
// ---------------------------------------------------------------------------
__global__ __launch_bounds__(256) void proj_kernel(const _Float16* __restrict__ Wh,
                                                   const _Float16* __restrict__ xT,
                                                   const float* __restrict__ bq,
                                                   const float* __restrict__ bk,
                                                   const float* __restrict__ bv,
                                                   _Float16* __restrict__ Qf,
                                                   _Float16* __restrict__ Kf,
                                                   _Float16* __restrict__ Vh) {
    int lane = threadIdx.x & 31;
    int job  = blockIdx.x * 8 + (threadIdx.x >> 5);      // 12288 wave jobs
    int nt = (job & 31) * 64;                            // N tile base (seq)
    int mt = ((job >> 5) & 31) * 16;                     // M tile base (out ch)
    int b  = (job >> 10) & 3;
    int w  = job >> 12;                                  // 0=Q 1=K 2=V

    const _Float16* A  = Wh + (size_t)w * CCH * CCH + (size_t)mt * CCH;
    const _Float16* Bp = xT + (size_t)b * LEN * CCH;

    v8f acc0 = {}, acc1 = {}, acc2 = {}, acc3 = {};
    for (int c0 = 0; c0 < CCH; c0 += 32) {
        v16h af = load_afrag(A + c0, CCH, lane);
        v16h b0 = load_bfrag(Bp + (size_t)(nt +  0) * CCH + c0, CCH, lane);
        v16h b1 = load_bfrag(Bp + (size_t)(nt + 16) * CCH + c0, CCH, lane);
        v16h b2 = load_bfrag(Bp + (size_t)(nt + 32) * CCH + c0, CCH, lane);
        v16h b3 = load_bfrag(Bp + (size_t)(nt + 48) * CCH + c0, CCH, lane);
        acc0 = wmma16(af, b0, acc0);
        acc1 = wmma16(af, b1, acc1);
        acc2 = wmma16(af, b2, acc2);
        acc3 = wmma16(af, b3, acc3);
    }

    const float* bias = (w == 0) ? bq : (w == 1) ? bk : bv;
    int lo = lane & 15, hi = lane >> 4;
    v8f* accs[4] = {&acc0, &acc1, &acc2, &acc3};

    for (int j = 0; j < 4; ++j) {
        for (int r = 0; r < 8; ++r) {
            int o = mt + r + 8 * hi;                     // output channel
            int l = nt + j * 16 + lo;                    // sequence position
            float val = (*accs[j])[r] + bias[o];
            if (w == 2) {
                Vh[((size_t)(b * CCH + o) << 11) + l] = (_Float16)val;
            } else {
                int h = o >> 6, d = o & 63;
                _Float16* Dst = (w == 0) ? Qf : Kf;
                Dst[(((size_t)(b * HEADS + h) * LEN + l) << 6) + d] = (_Float16)val;
            }
        }
    }
}

// ---------------------------------------------------------------------------
// Kernel 2: flash attention + fused  out = gamma * attn_out + x.
// One wave = 16 queries of one (b,h); key tiles of 32; online softmax with
// per-lane (per-query) stats (transposed S so N = queries).
// ---------------------------------------------------------------------------
__global__ __launch_bounds__(256) void attn_kernel(const _Float16* __restrict__ Qf,
                                                   const _Float16* __restrict__ Kf,
                                                   const _Float16* __restrict__ Vh,
                                                   const float* __restrict__ x,
                                                   const float* __restrict__ gamma,
                                                   float* __restrict__ out) {
    __shared__ _Float16 smem[8 * 16 * 32];               // 8 waves x Pt[16q][32k]
    int wid  = threadIdx.x >> 5;
    int lane = threadIdx.x & 31;
    int g  = blockIdx.x * 8 + wid;                       // 4096 wave jobs
    int qt = g & 127;
    int h  = (g >> 7) & 7;
    int b  = g >> 10;
    int l0 = qt * 16;
    int lo = lane & 15, hi = lane >> 4;

    const _Float16* Qb = Qf + ((size_t)(b * HEADS + h) * LEN) * DH;
    const _Float16* Kb = Kf + ((size_t)(b * HEADS + h) * LEN) * DH;
    const _Float16* Vb = Vh + ((size_t)(b * CCH + h * DH)) * LEN;
    _Float16* P = smem + wid * 512;

    // Q as B-fragments (column n = query, K run over d) — loop invariant
    v16h qb0 = load_bfrag(Qb + (size_t)l0 * DH +  0, DH, lane);
    v16h qb1 = load_bfrag(Qb + (size_t)l0 * DH + 32, DH, lane);

    v8f o0 = {}, o1 = {}, o2 = {}, o3 = {};
    float mrow = -1e30f, lrow = 0.0f;
    const float scale = 0.125f;                          // 1/sqrt(64)

    for (int m0 = 0; m0 < LEN; m0 += 32) {
        if (m0 + 32 < LEN) {                             // hint next K/V tiles
            __builtin_prefetch(Kb + (size_t)(m0 + 32) * DH, 0, 1);
            __builtin_prefetch(Vb + (size_t)m0 + 32, 0, 1);
        }
        // S^T = K_tile * Q^T  (M = keys, N = queries), K-dim = d = 64
        v8f z = {};
        v16h ka;
        ka = load_afrag(Kb + (size_t)m0 * DH +  0, DH, lane);
        v8f sA = wmma16(ka, qb0, z);
        ka = load_afrag(Kb + (size_t)m0 * DH + 32, DH, lane);
        sA = wmma16(ka, qb1, sA);
        ka = load_afrag(Kb + (size_t)(m0 + 16) * DH +  0, DH, lane);
        v8f sB = wmma16(ka, qb0, z);
        ka = load_afrag(Kb + (size_t)(m0 + 16) * DH + 32, DH, lane);
        sB = wmma16(ka, qb1, sB);

        // online softmax: stats per query = per lane (halves combined via xor16)
        float tmax = -1e30f;
        for (int r = 0; r < 8; ++r) {
            sA[r] *= scale; sB[r] *= scale;
            tmax = fmaxf(tmax, fmaxf(sA[r], sB[r]));
        }
        tmax = fmaxf(tmax, __shfl_xor(tmax, 16, 32));
        float mnew  = fmaxf(mrow, tmax);
        float alpha = __expf(mrow - mnew);
        v8h pA, pB;
        float ps = 0.0f;
        for (int r = 0; r < 8; ++r) {
            float eA = __expf(sA[r] - mnew);
            float eB = __expf(sB[r] - mnew);
            ps += eA + eB;
            pA[r] = (_Float16)eA;
            pB[r] = (_Float16)eB;
        }
        ps += __shfl_xor(ps, 16, 32);
        lrow = lrow * alpha + ps;
        mrow = mnew;
        for (int r = 0; r < 8; ++r) {
            o0[r] *= alpha; o1[r] *= alpha; o2[r] *= alpha; o3[r] *= alpha;
        }

        // P^T (keys x queries) -> LDS as [query][key] (key-contiguous per query)
        *(v8h*)(P + lo * 32 + hi * 8)      = pA;         // keys [8*hi, 8*hi+8)
        *(v8h*)(P + lo * 32 + 16 + hi * 8) = pB;         // keys 16 + [8*hi, ...)
        asm volatile("s_wait_dscnt 0x0" ::: "memory");

        // O = V_tile * P^T : A = V rows(d) x keys (l-contiguous), B = P from LDS
        v16h pb = load_bfrag(P, 32, lane);
        v16h va;
        va = load_afrag(Vb + (size_t) 0 * LEN + m0, LEN, lane); o0 = wmma16(va, pb, o0);
        va = load_afrag(Vb + (size_t)16 * LEN + m0, LEN, lane); o1 = wmma16(va, pb, o1);
        va = load_afrag(Vb + (size_t)32 * LEN + m0, LEN, lane); o2 = wmma16(va, pb, o2);
        va = load_afrag(Vb + (size_t)48 * LEN + m0, LEN, lane); o3 = wmma16(va, pb, o3);
    }

    // normalize + fused residual:  out = gamma * (O / l) + x
    float inv = 1.0f / lrow;
    float g0  = gamma[0];
    v8f* oacc[4] = {&o0, &o1, &o2, &o3};
    for (int dd = 0; dd < 4; ++dd) {
        for (int r = 0; r < 8; ++r) {
            int dl = dd * 16 + r + 8 * hi;               // d within head
            size_t idx = (((size_t)(b * CCH + h * DH + dl)) << 11) + l0 + lo;
            out[idx] = g0 * ((*oacc[dd])[r] * inv) + x[idx];
        }
    }
}

// ---------------------------------------------------------------------------
extern "C" void kernel_launch(void* const* d_in, const int* in_sizes, int n_in,
                              void* d_out, int out_size, void* d_ws, size_t ws_size,
                              hipStream_t stream) {
    (void)in_sizes; (void)n_in; (void)out_size; (void)ws_size;
    const float* x     = (const float*)d_in[0];
    const float* Wq    = (const float*)d_in[1];
    const float* bq    = (const float*)d_in[2];
    const float* Wk    = (const float*)d_in[3];
    const float* bk    = (const float*)d_in[4];
    const float* Wv    = (const float*)d_in[5];
    const float* bv    = (const float*)d_in[6];
    const float* gamma = (const float*)d_in[7];
    float* out = (float*)d_out;

    char* ws = (char*)d_ws;
    _Float16* xT = (_Float16*)(ws + OFF_XT);
    _Float16* Wh = (_Float16*)(ws + OFF_WH);
    _Float16* Qf = (_Float16*)(ws + OFF_QF);
    _Float16* Kf = (_Float16*)(ws + OFF_KF);
    _Float16* Vh = (_Float16*)(ws + OFF_VH);

    cvt_x_kernel<<<(BATCH * CCH * LEN) / 256, 256, 0, stream>>>(x, xT);
    cvt_w_kernel<<<(3 * CCH * CCH) / 256, 256, 0, stream>>>(Wq, Wk, Wv, Wh);
    // 3 mats * 4 batches * 32 Mtiles * 32 Ntiles(64-wide) = 12288 waves / 8 per block
    proj_kernel<<<1536, 256, 0, stream>>>(Wh, xT, bq, bk, bv, Qf, Kf, Vh);
    // 4*8*(2048/16) = 4096 waves / 8 per block
    attn_kernel<<<512, 256, 0, stream>>>(Qf, Kf, Vh, x, gamma, out);
}
